// Filterformer_6777458393703
// MI455X (gfx1250) — matrix operand order repaired and checked by reading, not compiled, not run on hardware
//
#include <hip/hip_runtime.h>

typedef __attribute__((ext_vector_type(16))) __bf16 v16bf;
typedef __attribute__((ext_vector_type(8)))  __bf16 v8bf;
typedef __attribute__((ext_vector_type(8)))  float  v8f;

#define B_DIM 32
#define T_DIM 96
#define O_DIM 32
#define BT    (B_DIM * T_DIM)   // 3072
#define NREF  256
#define KPATH 100
#define NSIM  128
#define NPOS  128
#define NTIME 64
#define ENCD  512
#define H1D   1024
#define H2D   1024
#define NGEO  256
#define NSTATE 64
#define FEATK (NSIM * O_DIM)    // 4096

// native f32 -> bf16 (RTE) : lowers to v_cvt_pk_bf16_f32 on gfx1250
static __device__ __forceinline__ __bf16 f2b(float f) {
  return static_cast<__bf16>(f);
}

static __device__ __forceinline__ v8f wmma_bf16(const v16bf& a, const v16bf& b, const v8f& c) {
  return __builtin_amdgcn_wmma_f32_16x16x32_bf16(false, a, false, b, (short)0, c, false, false);
}

// ---------------------------------------------------------------------------
// Weight convert + transpose: out[n*K + k] = bf16(in[k*N + n])
// ---------------------------------------------------------------------------
__global__ void wconv_kernel(const float* __restrict__ in, __bf16* __restrict__ out,
                             int K, int N, long total) {
  long i = (long)blockIdx.x * blockDim.x + threadIdx.x;
  if (i >= total) return;
  int n = (int)(i / K);
  int k = (int)(i % K);
  out[i] = f2b(in[(long)k * N + n]);
}

// ---------------------------------------------------------------------------
// dists[bt, n] = max_k sqrt(max(|Yi[bt,k,:]|^2 + |ref[n,k,:]|^2 - 2 dot, eps))
// One wave per 16x16 (bt x ref) tile; K-dim = O = 32 = one bf16 WMMA.
// ---------------------------------------------------------------------------
__global__ __launch_bounds__(32) void dist_kernel(const float* __restrict__ Y,
                                                  const float* __restrict__ ref,
                                                  float* __restrict__ dists) {
  __shared__ float Abuf[16 * 32];
  __shared__ float Bbuf[16 * 32];
  __shared__ float y2s[16];
  __shared__ float r2s[16];
  const int lane  = threadIdx.x;
  const int mBase = blockIdx.x * 16;
  const int nBase = blockIdx.y * 16;
  const int half  = lane >> 4;
  const int l15   = lane & 15;

  float acc[8];
#pragma unroll
  for (int i = 0; i < 8; ++i) acc[i] = 0.0f;

  for (int kk = 0; kk < KPATH; ++kk) {
    // stage interpolated Yi tile (16 bt-rows x 32 obs) and ref tile
#pragma unroll
    for (int j = 0; j < 16; ++j) {
      int e = lane * 16 + j;
      int row = e >> 5, o = e & 31;
      int bt = mBase + row;
      int t = bt % T_DIM, b = bt / T_DIM;
      float pos = (float)kk * (float)t * (1.0f / (KPATH - 1));
      int lo = (int)floorf(pos);
      int hi = lo + 1; if (hi > t) hi = t;
      float fr = pos - (float)lo;
      const float* yb = Y + ((long)b * T_DIM) * O_DIM;
      Abuf[e] = yb[lo * O_DIM + o] * (1.0f - fr) + yb[hi * O_DIM + o] * fr;
      Bbuf[e] = ref[((long)(nBase + row)) * (KPATH * O_DIM) + kk * O_DIM + o];
    }
    __syncthreads();
    {
      const float* src = half ? Bbuf : Abuf;
      float s = 0.f;
#pragma unroll
      for (int o = 0; o < 32; ++o) { float v = src[l15 * 32 + o]; s += v * v; }
      if (half) r2s[l15] = s; else y2s[l15] = s;
    }
    __syncthreads();
    // pack A (16x32) / B (32x16) with k = 16*half + j in element j of both (contiguous order)
    v16bf a, bm;
#pragma unroll
    for (int j = 0; j < 8; ++j) {
      a[j]     = f2b(Abuf[l15 * 32 + half * 16 + j]);
      a[j + 8] = f2b(Abuf[l15 * 32 + half * 16 + 8 + j]);
    }
#pragma unroll
    for (int j = 0; j < 16; ++j) bm[j] = f2b(Bbuf[l15 * 32 + half * 16 + j]);

    v8f cz = {};
    v8f d = wmma_bf16(a, bm, cz);
#pragma unroll
    for (int i = 0; i < 8; ++i) {
      int m = i + half * 8;
      float dsq = y2s[m] + r2s[l15] - 2.0f * d[i];
      float dd = __builtin_amdgcn_sqrtf(fmaxf(dsq, 1e-12f));
      acc[i] = fmaxf(acc[i], dd);
    }
    __syncthreads();
  }
#pragma unroll
  for (int i = 0; i < 8; ++i) {
    int m = i + half * 8;
    dists[((long)(mBase + m)) * NREF + nBase + l15] = acc[i];
  }
}

// ---------------------------------------------------------------------------
// sim = softmax(relu(dists @ Bsim^T) @ Asim^T + bsim) + asim   (tiny, VALU)
// ---------------------------------------------------------------------------
__global__ __launch_bounds__(128) void sim_kernel(const float* __restrict__ dists,
                                                  const float* __restrict__ Bsim,
                                                  const float* __restrict__ Asim,
                                                  const float* __restrict__ asim,
                                                  const float* __restrict__ bsim,
                                                  float* __restrict__ sim) {
  __shared__ float drow[NREF];
  __shared__ float tmp[NSIM];
  __shared__ float red[NSIM];
  const int bt = blockIdx.x;
  const int tid = threadIdx.x;
  drow[tid]       = dists[(long)bt * NREF + tid];
  drow[tid + 128] = dists[(long)bt * NREF + tid + 128];
  __syncthreads();
  float s = 0.f;
  for (int n = 0; n < NREF; ++n) s += Bsim[(long)tid * NREF + n] * drow[n];
  tmp[tid] = fmaxf(s, 0.f);
  __syncthreads();
  float inner = bsim[tid];
  for (int k = 0; k < NSIM; ++k) inner += Asim[(long)tid * NSIM + k] * tmp[k];
  red[tid] = inner; __syncthreads();
  for (int st = 64; st > 0; st >>= 1) { if (tid < st) red[tid] = fmaxf(red[tid], red[tid + st]); __syncthreads(); }
  float mx = red[0]; __syncthreads();
  float e = __expf(inner - mx);
  red[tid] = e; __syncthreads();
  for (int st = 64; st > 0; st >>= 1) { if (tid < st) red[tid] += red[tid + st]; __syncthreads(); }
  float sum = red[0];
  sim[(long)bt * NSIM + tid] = e / sum + asim[tid];
}

// ---------------------------------------------------------------------------
// feat[bt, p*32+o] = sim[bt,p] * (sum_m Upos[p,m]*Y[b,tq[t,m],o] + Vpos[p,o])
// per-bt 128x32x64 GEMM -> 8x2 tiles x 2 WMMA k-steps, bf16 output.
// ---------------------------------------------------------------------------
__global__ __launch_bounds__(32) void feat_kernel(const float* __restrict__ Y,
                                                  const float* __restrict__ times,
                                                  const float* __restrict__ Upos,
                                                  const float* __restrict__ Vpos,
                                                  const float* __restrict__ sim,
                                                  __bf16* __restrict__ feat) {
  __shared__ __bf16 ysb[NTIME * O_DIM];
  const int bt = blockIdx.x;
  const int t = bt % T_DIM, b = bt / T_DIM;
  const int lane = threadIdx.x;
  const int half = lane >> 4;
  const int l15  = lane & 15;
  // gather ys (64 x 32) into LDS as bf16
  for (int r = 0; r < 2; ++r) {
    int m = lane * 2 + r;
    int tq = (int)floorf(times[m] * (float)t);
    if (tq < 0) tq = 0;
    if (tq > t) tq = t;
    const float* yr = Y + ((long)b * T_DIM + tq) * O_DIM;
#pragma unroll
    for (int o = 0; o < O_DIM; ++o) ysb[m * O_DIM + o] = f2b(yr[o]);
  }
  __syncthreads();
  for (int pt = 0; pt < 8; ++pt) {
    int pBase = pt * 16;
    for (int ot = 0; ot < 2; ++ot) {
      int oBase = ot * 16;
      v8f c = {};
      for (int ks = 0; ks < NTIME; ks += 32) {
        v16bf a, bm;
        const float* up = Upos + (long)(pBase + l15) * NTIME;
#pragma unroll
        for (int j = 0; j < 8; ++j) {
          a[j]     = f2b(up[ks + half * 16 + j]);
          a[j + 8] = f2b(up[ks + half * 16 + 8 + j]);
        }
#pragma unroll
        for (int j = 0; j < 16; ++j)
          bm[j] = ysb[(ks + half * 16 + j) * O_DIM + oBase + l15];
        c = wmma_bf16(a, bm, c);
      }
#pragma unroll
      for (int i = 0; i < 8; ++i) {
        int p = pBase + i + half * 8;
        int o = oBase + l15;
        float val = c[i] + Vpos[(long)p * O_DIM + o];
        float fe = sim[(long)bt * NSIM + p] * val;
        feat[(long)bt * FEATK + p * O_DIM + o] = f2b(fe);
      }
    }
  }
}

// ---------------------------------------------------------------------------
// Generic bf16 WMMA GEMM with 2x2 register blocking:
//   out = act(A(MxK) @ Wt^T(NxK) + bias [+ tval*tadd])
// 4 waves/block; each wave owns a 32x32 output tile (4 WMMA per k-step,
// 2 A + 2 B fetches). 16B vector loads put k = 16*half + s in element s of A and B alike.
// ---------------------------------------------------------------------------
__global__ __launch_bounds__(128) void gemm_kernel(const __bf16* __restrict__ A,
                                                   const __bf16* __restrict__ Wt,
                                                   const float* __restrict__ bias,
                                                   const float* __restrict__ tadd,
                                                   __bf16* __restrict__ outB,
                                                   float* __restrict__ outF,
                                                   int N, int K, int doRelu) {
  const int lane  = threadIdx.x & 31;
  const int wave  = threadIdx.x >> 5;
  const int mBase = blockIdx.x * 32;
  const int nBase = blockIdx.y * 128 + wave * 32;
  const int half  = lane >> 4;
  const int l15   = lane & 15;
  const __bf16* arow0 = A  + (long)(mBase + l15) * K;
  const __bf16* arow1 = arow0 + 16L * K;
  const __bf16* wrow0 = Wt + (long)(nBase + l15) * K;
  const __bf16* wrow1 = wrow0 + 16L * K;
  v8f c00 = {}, c01 = {}, c10 = {}, c11 = {};
  for (int k = 0; k < K; k += 32) {
    if (k + 256 < K) {
      __builtin_prefetch(arow0 + k + 256, 0, 0);
      __builtin_prefetch(wrow0 + k + 256, 0, 0);
    }
    v8bf a00 = *reinterpret_cast<const v8bf*>(arow0 + k + half * 16);
    v8bf a01 = *reinterpret_cast<const v8bf*>(arow0 + k + half * 16 + 8);
    v8bf a10 = *reinterpret_cast<const v8bf*>(arow1 + k + half * 16);
    v8bf a11 = *reinterpret_cast<const v8bf*>(arow1 + k + half * 16 + 8);
    v8bf b00 = *reinterpret_cast<const v8bf*>(wrow0 + k + half * 16);
    v8bf b01 = *reinterpret_cast<const v8bf*>(wrow0 + k + half * 16 + 8);
    v8bf b10 = *reinterpret_cast<const v8bf*>(wrow1 + k + half * 16);
    v8bf b11 = *reinterpret_cast<const v8bf*>(wrow1 + k + half * 16 + 8);
    v16bf a0, a1, b0, b1;
#pragma unroll
    for (int j = 0; j < 8; ++j) {
      a0[j] = a00[j]; a0[j + 8] = a01[j];
      a1[j] = a10[j]; a1[j + 8] = a11[j];
      b0[j] = b00[j]; b0[j + 8] = b01[j];
      b1[j] = b10[j]; b1[j + 8] = b11[j];
    }
    c00 = wmma_bf16(a0, b0, c00);
    c01 = wmma_bf16(a0, b1, c01);
    c10 = wmma_bf16(a1, b0, c10);
    c11 = wmma_bf16(a1, b1, c11);
  }
#pragma unroll
  for (int tm = 0; tm < 2; ++tm) {
#pragma unroll
    for (int tn = 0; tn < 2; ++tn) {
      const v8f& c = tm ? (tn ? c11 : c10) : (tn ? c01 : c00);
#pragma unroll
      for (int i = 0; i < 8; ++i) {
        int m = mBase + tm * 16 + i + half * 8;
        int n = nBase + tn * 16 + l15;
        float v = c[i] + bias[n];
        if (tadd) v += (float)((m % T_DIM) + 1) * tadd[n];   // encoder tval column
        if (doRelu) v = fmaxf(v, 0.f);
        if (outB) outB[(long)m * N + n] = f2b(v);
        if (outF) outF[(long)m * N + n] = v;
      }
    }
  }
}

// ---------------------------------------------------------------------------
// out[bt,:] = softmax(logits[bt,:]) @ geo_means
// ---------------------------------------------------------------------------
__global__ __launch_bounds__(256) void geo_kernel(const float* __restrict__ logits,
                                                  const float* __restrict__ geoM,
                                                  float* __restrict__ out) {
  __shared__ float w[NGEO];
  __shared__ float red[NGEO];
  const int bt = blockIdx.x;
  const int tid = threadIdx.x;
  float lg = logits[(long)bt * NGEO + tid];
  red[tid] = lg; __syncthreads();
  for (int st = 128; st > 0; st >>= 1) { if (tid < st) red[tid] = fmaxf(red[tid], red[tid + st]); __syncthreads(); }
  float mx = red[0]; __syncthreads();
  float e = __expf(lg - mx);
  red[tid] = e; __syncthreads();
  for (int st = 128; st > 0; st >>= 1) { if (tid < st) red[tid] += red[tid + st]; __syncthreads(); }
  float sum = red[0]; __syncthreads();
  w[tid] = e / sum;
  __syncthreads();
  if (tid < NSTATE) {
    float s = 0.f;
    for (int g = 0; g < NGEO; ++g) s += w[g] * geoM[(long)g * NSTATE + tid];
    out[(long)bt * NSTATE + tid] = s;
  }
}

extern "C" void kernel_launch(void* const* d_in, const int* in_sizes, int n_in,
                              void* d_out, int out_size, void* d_ws, size_t ws_size,
                              hipStream_t stream) {
  (void)in_sizes; (void)n_in; (void)out_size; (void)ws_size;
  const float* Y     = (const float*)d_in[0];
  const float* ref   = (const float*)d_in[1];
  const float* Asim  = (const float*)d_in[2];
  const float* Bsim  = (const float*)d_in[3];
  const float* asim  = (const float*)d_in[4];
  const float* bsim  = (const float*)d_in[5];
  const float* times = (const float*)d_in[6];
  const float* Upos  = (const float*)d_in[7];
  const float* Vpos  = (const float*)d_in[8];
  const float* CW    = (const float*)d_in[9];
  const float* Cb    = (const float*)d_in[10];
  const float* W1    = (const float*)d_in[11];
  const float* b1    = (const float*)d_in[12];
  const float* W2    = (const float*)d_in[13];
  const float* b2    = (const float*)d_in[14];
  const float* geoW  = (const float*)d_in[15];
  const float* geob  = (const float*)d_in[16];
  const float* geoM  = (const float*)d_in[17];
  float* out = (float*)d_out;

  char* ws = (char*)d_ws;
  size_t off = 0;
  auto alloc = [&](size_t bytes) -> char* {
    char* p = ws + off;
    off += (bytes + 255) & ~(size_t)255;
    return p;
  };
  float*  dists  = (float*)alloc((size_t)BT * NREF * 4);
  float*  sim    = (float*)alloc((size_t)BT * NSIM * 4);
  __bf16* feat   = (__bf16*)alloc((size_t)BT * FEATK * 2);
  __bf16* encB   = (__bf16*)alloc((size_t)BT * ENCD * 2);
  __bf16* h1B    = (__bf16*)alloc((size_t)BT * H1D * 2);
  __bf16* h2B    = (__bf16*)alloc((size_t)BT * H2D * 2);
  float*  logits = (float*)alloc((size_t)BT * NGEO * 4);
  __bf16* CWt    = (__bf16*)alloc((size_t)ENCD * FEATK * 2);
  __bf16* W1t    = (__bf16*)alloc((size_t)H1D * ENCD * 2);
  __bf16* W2t    = (__bf16*)alloc((size_t)H2D * H1D * 2);
  __bf16* geoWt  = (__bf16*)alloc((size_t)NGEO * H2D * 2);

  // weight prep (transpose + f32->bf16)
  { long tot = (long)ENCD * FEATK;
    wconv_kernel<<<dim3((tot + 255) / 256), dim3(256), 0, stream>>>(CW + ENCD, CWt, FEATK, ENCD, tot); }
  { long tot = (long)H1D * ENCD;
    wconv_kernel<<<dim3((tot + 255) / 256), dim3(256), 0, stream>>>(W1, W1t, ENCD, H1D, tot); }
  { long tot = (long)H2D * H1D;
    wconv_kernel<<<dim3((tot + 255) / 256), dim3(256), 0, stream>>>(W2, W2t, H1D, H2D, tot); }
  { long tot = (long)NGEO * H2D;
    wconv_kernel<<<dim3((tot + 255) / 256), dim3(256), 0, stream>>>(geoW, geoWt, H2D, NGEO, tot); }

  dist_kernel<<<dim3(BT / 16, NREF / 16), dim3(32), 0, stream>>>(Y, ref, dists);
  sim_kernel<<<dim3(BT), dim3(128), 0, stream>>>(dists, Bsim, Asim, asim, bsim, sim);
  feat_kernel<<<dim3(BT), dim3(32), 0, stream>>>(Y, times, Upos, Vpos, sim, feat);

  gemm_kernel<<<dim3(BT / 32, ENCD / 128), dim3(128), 0, stream>>>(
      feat, CWt, Cb, CW /* row 0 */, encB, nullptr, ENCD, FEATK, 0);
  gemm_kernel<<<dim3(BT / 32, H1D / 128), dim3(128), 0, stream>>>(
      encB, W1t, b1, nullptr, h1B, nullptr, H1D, ENCD, 1);
  gemm_kernel<<<dim3(BT / 32, H2D / 128), dim3(128), 0, stream>>>(
      h1B, W2t, b2, nullptr, h2B, nullptr, H2D, H1D, 1);
  gemm_kernel<<<dim3(BT / 32, NGEO / 128), dim3(128), 0, stream>>>(
      h2B, geoWt, geob, nullptr, nullptr, logits, NGEO, H2D, 0);

  geo_kernel<<<dim3(BT), dim3(256), 0, stream>>>(logits, geoM, out);
}
